// Decoder_15702400434702
// MI455X (gfx1250) — matrix operand order repaired and checked by reading, not compiled, not run on hardware
//
#include <hip/hip_runtime.h>
#include <hip/hip_bf16.h>
#include <math.h>

#define B_ 128
#define S_ 512
#define D_ 512
#define H_ 512
#define V_ 50000

typedef __attribute__((ext_vector_type(16))) __bf16 v16bf;
typedef __attribute__((ext_vector_type(8)))  float  v8f;

union BF16x16 { v16bf v; unsigned short h[16]; };

__device__ __forceinline__ unsigned short f2bf(float f) {
  unsigned int u = __builtin_bit_cast(unsigned int, f);
  u += 0x7FFFu + ((u >> 16) & 1u);   // round-to-nearest-even
  return (unsigned short)(u >> 16);
}

__device__ __forceinline__ v8f wmma_bf16(v16bf a, v16bf b, v8f c) {
  return __builtin_amdgcn_wmma_f32_16x16x32_bf16(false, a, false, b, (short)0, c, false, false);
}

__device__ __forceinline__ float fast_tanh(float x) {
#if defined(__has_builtin)
#if __has_builtin(__builtin_amdgcn_tanhf)
  return __builtin_amdgcn_tanhf(x);
#elif __has_builtin(__builtin_amdgcn_tanh_f32)
  return __builtin_amdgcn_tanh_f32(x);
#else
  return tanhf(x);
#endif
#else
  return tanhf(x);
#endif
}

__device__ __forceinline__ float sigmf(float x) { return 1.f / (1.f + __expf(-x)); }

// ---- fragment builders (ISA 7.12.2 layouts, wave32) ----
// A (16x32 bf16): lane<16 -> M=lane, K {kb+0..7, kb+16..23}; lane>=16 -> M=lane-16, K {kb+8..15, kb+24..31}
__device__ __forceinline__ v16bf afrag_global(const float* __restrict__ M, int ld,
                                              int m0, int kb, int lane) {
  int row  = lane & 15;
  int half = (lane < 16) ? 0 : 8;
  const float* p = M + (size_t)(m0 + row) * ld + kb;
  BF16x16 r;
#pragma unroll
  for (int j = 0; j < 8; ++j) r.h[j]     = f2bf(p[half + j]);
#pragma unroll
  for (int j = 0; j < 8; ++j) r.h[8 + j] = f2bf(p[16 + half + j]);
  return r.v;
}

// B (32x16 bf16): lane<16 -> N=lane, K {kb+0..7, kb+16..23}; lane>=16 -> N=lane-16, K {kb+8..15, kb+24..31}
__device__ __forceinline__ v16bf bfrag_global(const float* __restrict__ M, int ld,
                                              int n0, int kb, int lane) {
  int col  = lane & 15;
  int koff = (lane < 16) ? 0 : 8;
  const float* p = M + (size_t)(n0 + col) * ld + kb + koff;
  BF16x16 r;
#pragma unroll
  for (int j = 0; j < 8; ++j) r.h[j] = f2bf(p[j]);
#pragma unroll
  for (int j = 0; j < 8; ++j) r.h[8 + j] = f2bf(p[16 + j]);
  return r.v;
}

__device__ __forceinline__ v16bf afrag_lds(const unsigned short* base, int ldu,
                                           int r0, int kb, int lane) {
  int row  = r0 + (lane & 15);
  int half = (lane < 16) ? 0 : 8;
  const unsigned short* p = base + row * ldu + kb;
  BF16x16 r;
#pragma unroll
  for (int j = 0; j < 8; ++j) r.h[j]     = p[half + j];
#pragma unroll
  for (int j = 0; j < 8; ++j) r.h[8 + j] = p[16 + half + j];
  return r.v;
}

// packed fragment buffers: frag index fi -> 32 lanes x 16 bf16 (32B per lane)
__device__ __forceinline__ v16bf frag_packed(const unsigned short* __restrict__ base,
                                             int fi, int lane) {
  return *(const v16bf*)(base + ((size_t)fi * 32 + lane) * 16);
}

// ---- 0a. pack A fragments of a [128 x 512] fp32 matrix (grid 8*16, block 32) ----
__global__ void pack_a_kernel(const float* __restrict__ src, unsigned short* __restrict__ dst) {
  int blk = blockIdx.x;          // blk = mt*16 + ks
  int ks = blk & 15, mt = blk >> 4;
  int lane = threadIdx.x;
  v16bf f = afrag_global(src, 512, mt * 16, ks * 32, lane);
  *(v16bf*)(dst + ((size_t)blk * 32 + lane) * 16) = f;
}

// ---- 0b. pack B fragments of W_ctx [512 x 512] (grid 32*16, block 32) ----
__global__ void pack_b_kernel(const float* __restrict__ src, unsigned short* __restrict__ dst) {
  int blk = blockIdx.x;          // blk = et*16 + ks
  int ks = blk & 15, et = blk >> 4;
  int lane = threadIdx.x;
  v16bf f = bfrag_global(src, 512, et * 16, ks * 32, lane);
  *(v16bf*)(dst + ((size_t)blk * 32 + lane) * 16) = f;
}

// ---- 1. wq = h0 @ W_q^T + b_q  [B,D] ----
__global__ void wq_kernel(const float* __restrict__ h0, const float* __restrict__ W_q,
                          const float* __restrict__ b_q, float* __restrict__ wq) {
  __shared__ float hrow[D_];
  int b = blockIdx.x;
  for (int d = threadIdx.x; d < D_; d += blockDim.x) hrow[d] = h0[b * D_ + d];
  __syncthreads();
  for (int e = threadIdx.x; e < D_; e += blockDim.x) {
    const float* w = W_q + (size_t)e * D_;
    float acc = b_q[e];
    for (int d = 0; d < D_; ++d) acc = fmaf(hrow[d], w[d], acc);
    wq[b * D_ + e] = acc;
  }
}

// ---- 2. fused scores[b,s] = v . tanh(wq[b] + enc_hs[b,s,:] @ W_ctx^T) + v_b ----
#define AT_LDU 528  // padded ushort row stride to spread LDS banks
__global__ void attn_scores_kernel(const float* __restrict__ enc,
                                   const unsigned short* __restrict__ pk_wctx,
                                   const float* __restrict__ wq,
                                   const float* __restrict__ v_w,
                                   const float* __restrict__ v_b,
                                   float* __restrict__ scores) {
  __shared__ unsigned short enc_s[32 * AT_LDU];   // 32 s-rows as bf16
  int blk = blockIdx.x;
  int b = blk >> 4;            // / (S/32)
  int st = blk & 15;
  int s0 = st * 32;
  int tid = threadIdx.x;       // 0..63 (2 waves)
  int w = tid >> 5;
  int lane = tid & 31;
  int col = lane & 15;

  for (int idx = tid; idx < 32 * D_; idx += 64) {
    int r = idx >> 9, k = idx & 511;
    enc_s[r * AT_LDU + k] = f2bf(enc[(size_t)(b * S_ + s0 + r) * D_ + k]);
  }
  __syncthreads();

  float acc[8];
#pragma unroll
  for (int i = 0; i < 8; ++i) acc[i] = 0.f;

  for (int et = 0; et < 32; ++et) {
    int e0 = et * 16;
    v8f c = (v8f)0.0f;
#pragma unroll 4
    for (int ks = 0; ks < 16; ++ks) {
      v16bf a  = afrag_lds(enc_s, AT_LDU, w * 16, ks * 32, lane);
      v16bf bb = frag_packed(pk_wctx, et * 16 + ks, lane);
      c = wmma_bf16(a, bb, c);
    }
    float wqv = wq[b * D_ + e0 + col];
    float vv  = v_w[e0 + col];
#pragma unroll
    for (int i = 0; i < 8; ++i) acc[i] += fast_tanh(c[i] + wqv) * vv;
  }

  // reduce the 16 e-columns held across each 16-lane half
#pragma unroll
  for (int i = 0; i < 8; ++i) {
    float a = acc[i];
    a += __shfl_xor(a, 1, 32);
    a += __shfl_xor(a, 2, 32);
    a += __shfl_xor(a, 4, 32);
    a += __shfl_xor(a, 8, 32);
    acc[i] = a;
  }
  if (col == 0) {
    float vb = v_b[0];
    int rbase = s0 + w * 16 + (lane >> 4) * 8;
#pragma unroll
    for (int i = 0; i < 8; ++i) scores[b * S_ + rbase + i] = acc[i] + vb;
  }
}

// ---- 3. masked softmax + ctx = attn @ enc_hs ----
__global__ void softmax_ctx_kernel(const float* __restrict__ scores,
                                   const int* __restrict__ mask,
                                   const float* __restrict__ enc,
                                   float* __restrict__ attn_out,
                                   float* __restrict__ ctx) {
  __shared__ float sc[S_];
  __shared__ float red[256];
  int b = blockIdx.x;
  int tid = threadIdx.x;
  float ninf = -__builtin_inff();
  float lmax = ninf;
  for (int s = tid; s < S_; s += 256) {
    float v = scores[b * S_ + s];
    v = mask[b * S_ + s] ? ninf : v;   // reference: where(src_mask, -inf, scores)
    sc[s] = v;
    lmax = fmaxf(lmax, v);
  }
  red[tid] = lmax;
  __syncthreads();
  for (int off = 128; off > 0; off >>= 1) {
    if (tid < off) red[tid] = fmaxf(red[tid], red[tid + off]);
    __syncthreads();
  }
  float m = red[0];
  __syncthreads();
  float lsum = 0.f;
  for (int s = tid; s < S_; s += 256) {
    float e = expf(sc[s] - m);
    sc[s] = e;
    lsum += e;
  }
  red[tid] = lsum;
  __syncthreads();
  for (int off = 128; off > 0; off >>= 1) {
    if (tid < off) red[tid] += red[tid + off];
    __syncthreads();
  }
  float inv = 1.f / red[0];
  __syncthreads();
  for (int s = tid; s < S_; s += 256) {
    float a = sc[s] * inv;
    sc[s] = a;
    attn_out[b * S_ + s] = a;
  }
  __syncthreads();
  for (int d = tid; d < D_; d += 256) {
    float acc = 0.f;
    for (int s = 0; s < S_; ++s)
      acc = fmaf(sc[s], enc[(size_t)(b * S_ + s) * D_ + d], acc);
    ctx[b * D_ + d] = acc;
  }
}

// ---- 4. gates = [y_prev|ctx] @ W_ih^T + h0 @ W_hh^T + biases  [B,4H] ----
// one wave per 16-wide N tile, all 8 M tiles resident -> weights read exactly once
__global__ void gates_kernel(const unsigned short* __restrict__ pk_yprev,
                             const unsigned short* __restrict__ pk_ctx,
                             const unsigned short* __restrict__ pk_h0,
                             const float* __restrict__ W_ih,
                             const float* __restrict__ W_hh,
                             const float* __restrict__ b_ih,
                             const float* __restrict__ b_hh,
                             float* __restrict__ gates) {
  int n0 = blockIdx.x * 16;
  int lane = threadIdx.x;
  int col = lane & 15;
  v8f c[8];
#pragma unroll
  for (int mt = 0; mt < 8; ++mt) c[mt] = (v8f)0.0f;

  for (int ks = 0; ks < 32; ++ks) {          // K over s_cur = [y_prev | ctx], 1024
    int kb = ks * 32;
    v16bf bb = bfrag_global(W_ih, 2 * D_, n0, kb, lane);
    const unsigned short* pk = (ks < 16) ? pk_yprev : pk_ctx;
    int ks2 = ks & 15;
#pragma unroll
    for (int mt = 0; mt < 8; ++mt) {
      v16bf a = frag_packed(pk, mt * 16 + ks2, lane);
      c[mt] = wmma_bf16(a, bb, c[mt]);
    }
  }
  for (int ks = 0; ks < 16; ++ks) {          // K over h0, 512
    v16bf bb = bfrag_global(W_hh, H_, n0, ks * 32, lane);
#pragma unroll
    for (int mt = 0; mt < 8; ++mt) {
      v16bf a = frag_packed(pk_h0, mt * 16 + ks, lane);
      c[mt] = wmma_bf16(a, bb, c[mt]);
    }
  }
  float bias = b_ih[n0 + col] + b_hh[n0 + col];
  int rhalf = (lane < 16) ? 0 : 8;
#pragma unroll
  for (int mt = 0; mt < 8; ++mt)
#pragma unroll
    for (int i = 0; i < 8; ++i)
      gates[(size_t)(mt * 16 + i + rhalf) * (4 * H_) + n0 + col] = c[mt][i] + bias;
}

// ---- 5. LSTM cell/hidden elementwise ----
__global__ void lstm_elem_kernel(const float* __restrict__ gates,
                                 const float* __restrict__ c0,
                                 float* __restrict__ hidden_out,
                                 float* __restrict__ cell_out) {
  int b = blockIdx.x;
  const float* g = gates + (size_t)b * 4 * H_;
  for (int h = threadIdx.x; h < H_; h += blockDim.x) {
    float ig = g[h], fg = g[H_ + h], gg = g[2 * H_ + h], og = g[3 * H_ + h];
    float cell = sigmf(fg) * c0[b * H_ + h] + sigmf(ig) * fast_tanh(gg);
    float hid  = sigmf(og) * fast_tanh(cell);
    cell_out[b * H_ + h]   = cell;
    hidden_out[b * H_ + h] = hid;
  }
}

// ---- 6. output = hidden @ ent_W^T + ent_b  [B,V] ----
__global__ void proj_kernel(const unsigned short* __restrict__ pk_hidden,
                            const float* __restrict__ ent_W,
                            const float* __restrict__ ent_b,
                            float* __restrict__ out) {
  int v0 = blockIdx.x * 16;
  int lane = threadIdx.x;
  int col = lane & 15;
  v8f c[8];
#pragma unroll
  for (int mt = 0; mt < 8; ++mt) c[mt] = (v8f)0.0f;

  const float* prow = ent_W + (size_t)(v0 + col) * H_;  // this lane's weight row
  for (int ks = 0; ks < 16; ++ks) {
    int kb = ks * 32;
    __builtin_prefetch(prow + kb + 128, 0, 1);          // global_prefetch_b8, stream ahead
    v16bf bb = bfrag_global(ent_W, H_, v0, kb, lane);
#pragma unroll
    for (int mt = 0; mt < 8; ++mt) {
      v16bf a = frag_packed(pk_hidden, mt * 16 + ks, lane);
      c[mt] = wmma_bf16(a, bb, c[mt]);
    }
  }
  float bias = ent_b[v0 + col];
  int rhalf = (lane < 16) ? 0 : 8;
#pragma unroll
  for (int mt = 0; mt < 8; ++mt)
#pragma unroll
    for (int i = 0; i < 8; ++i)
      out[(size_t)(mt * 16 + i + rhalf) * V_ + v0 + col] = c[mt][i] + bias;
}

extern "C" void kernel_launch(void* const* d_in, const int* in_sizes, int n_in,
                              void* d_out, int out_size, void* d_ws, size_t ws_size,
                              hipStream_t stream) {
  const float* y_prev = (const float*)d_in[0];
  const float* h0     = (const float*)d_in[1];
  const float* c0     = (const float*)d_in[2];
  const float* enc_hs = (const float*)d_in[3];
  // d_in[4] src_word_embeds: unused by reference
  const int*   src_mask = (const int*)d_in[5];
  const float* W_ctx  = (const float*)d_in[6];
  const float* W_q    = (const float*)d_in[7];
  const float* b_q    = (const float*)d_in[8];
  const float* v_w    = (const float*)d_in[9];
  const float* v_b    = (const float*)d_in[10];
  const float* W_ih   = (const float*)d_in[11];
  const float* W_hh   = (const float*)d_in[12];
  const float* b_ih   = (const float*)d_in[13];
  const float* b_hh   = (const float*)d_in[14];
  const float* ent_W  = (const float*)d_in[15];
  const float* ent_b  = (const float*)d_in[16];

  float* out        = (float*)d_out;                    // [B,V]
  float* out_hidden = out + (size_t)B_ * V_;            // [B,H]
  float* out_cell   = out_hidden + (size_t)B_ * H_;     // [B,H]
  float* out_attn   = out_cell + (size_t)B_ * H_;       // [B,S]

  float* ws     = (float*)d_ws;
  float* wq     = ws;                       // B*D
  float* scores = wq + (size_t)B_ * D_;     // B*S
  float* ctx    = scores + (size_t)B_ * S_; // B*D
  float* gates  = ctx + (size_t)B_ * D_;    // B*4H
  float* pkbase = gates + (size_t)B_ * 4 * H_;
  unsigned short* pk_yprev  = (unsigned short*)(pkbase);           // 128x512 bf16
  unsigned short* pk_ctx    = (unsigned short*)(pkbase + 32768);
  unsigned short* pk_h0     = (unsigned short*)(pkbase + 65536);
  unsigned short* pk_hidden = (unsigned short*)(pkbase + 98304);
  unsigned short* pk_wctx   = (unsigned short*)(pkbase + 131072);  // 512x512 bf16

  wq_kernel<<<B_, 256, 0, stream>>>(h0, W_q, b_q, wq);
  pack_b_kernel<<<32 * 16, 32, 0, stream>>>(W_ctx, pk_wctx);
  pack_a_kernel<<<8 * 16, 32, 0, stream>>>(y_prev, pk_yprev);
  pack_a_kernel<<<8 * 16, 32, 0, stream>>>(h0, pk_h0);
  attn_scores_kernel<<<B_ * (S_ / 32), 64, 0, stream>>>(enc_hs, pk_wctx, wq, v_w, v_b, scores);
  softmax_ctx_kernel<<<B_, 256, 0, stream>>>(scores, src_mask, enc_hs, out_attn, ctx);
  pack_a_kernel<<<8 * 16, 32, 0, stream>>>(ctx, pk_ctx);
  gates_kernel<<<(4 * H_) / 16, 32, 0, stream>>>(pk_yprev, pk_ctx, pk_h0, W_ih, W_hh,
                                                 b_ih, b_hh, gates);
  lstm_elem_kernel<<<B_, 256, 0, stream>>>(gates, c0, out_hidden, out_cell);
  pack_a_kernel<<<8 * 16, 32, 0, stream>>>(out_hidden, pk_hidden);
  proj_kernel<<<V_ / 16, 32, 0, stream>>>(pk_hidden, ent_W, ent_b, out);
}
